// RythmFinancialExpertMLP_51745765982796
// MI455X (gfx1250) — matrix operand order repaired and checked by reading, not compiled, not run on hardware
//
#include <hip/hip_runtime.h>
#include <hip/hip_bf16.h>
#include <cstdint>
#include <cstddef>

// ---------------------------------------------------------------------------
// MoE dense expert MLP for MI455X (gfx1250, wave32, WMMA bf16 16x16x32)
// BK=64 (128B-coalesced tile rows), ping-pong LDS double buffering
// (one barrier per K-step), GLOBAL_LOAD_ASYNC_TO_LDS_B128 copy path.
// ---------------------------------------------------------------------------

typedef __bf16 bf16;
typedef __attribute__((ext_vector_type(16))) __bf16 v16bf;
typedef __attribute__((ext_vector_type(8)))  __bf16 v8bf;
typedef __attribute__((ext_vector_type(4)))  __bf16 v4bf;
typedef __attribute__((ext_vector_type(8)))  float  v8f;
typedef __attribute__((ext_vector_type(4)))  int    i32x4;

static constexpr int T_TOK = 4096;   // B*S
static constexpr int HDIM  = 2048;
static constexpr int IDIM  = 4096;
static constexpr int NEXP  = 8;

// ---- async global->LDS copy path (gfx1250), guarded so either way compiles --
#if defined(__has_builtin)
# if __has_builtin(__builtin_amdgcn_global_load_async_to_lds_b128)
#  define USE_ASYNC_LDS 1
# endif
#endif
#ifndef USE_ASYNC_LDS
# define USE_ASYNC_LDS 0
#endif

#if USE_ASYNC_LDS
typedef __attribute__((address_space(1))) i32x4* gi4p;   // global int4*
typedef __attribute__((address_space(3))) i32x4* li4p;   // LDS int4*
#endif

__device__ __forceinline__ void cp16(bf16* ldst, const bf16* gsrc) {
#if USE_ASYNC_LDS
    __builtin_amdgcn_global_load_async_to_lds_b128(
        (gi4p)reinterpret_cast<i32x4*>(const_cast<bf16*>(gsrc)),
        (li4p)reinterpret_cast<i32x4*>(ldst), 0, 0);
#else
    *reinterpret_cast<v8bf*>(ldst) = *reinterpret_cast<const v8bf*>(gsrc);
#endif
}

__device__ __forceinline__ void cp_wait() {
#if USE_ASYNC_LDS
# if __has_builtin(__builtin_amdgcn_s_wait_asynccnt)
    __builtin_amdgcn_s_wait_asynccnt(0);
# else
    asm volatile("s_wait_asynccnt 0" ::: "memory");
# endif
#endif
}

__device__ __forceinline__ v16bf join16(v8bf lo, v8bf hi) {
    union { v8bf h[2]; v16bf v; } u;
    u.h[0] = lo; u.h[1] = hi;
    return u.v;
}

// ---------------------------------------------------------------------------
// Zero-out fp32 buffer (float4 stores)
// ---------------------------------------------------------------------------
__global__ __launch_bounds__(256) void zero_f32(float4* __restrict__ p, size_t n4) {
    size_t i = (size_t)blockIdx.x * blockDim.x + threadIdx.x;
    size_t stride = (size_t)gridDim.x * blockDim.x;
    float4 z; z.x = z.y = z.z = z.w = 0.f;
    for (; i < n4; i += stride) p[i] = z;
}

// ---------------------------------------------------------------------------
// fp32 -> bf16 convert (vectorized)
// ---------------------------------------------------------------------------
__global__ __launch_bounds__(256) void cvt_bf16(const float4* __restrict__ in,
                                                v4bf* __restrict__ out, size_t n4) {
    size_t i = (size_t)blockIdx.x * blockDim.x + threadIdx.x;
    size_t stride = (size_t)gridDim.x * blockDim.x;
    for (; i < n4; i += stride) {
        float4 v = in[i];
        v4bf o;
        o[0] = (bf16)v.x; o[1] = (bf16)v.y; o[2] = (bf16)v.z; o[3] = (bf16)v.w;
        out[i] = o;
    }
}

// ---------------------------------------------------------------------------
// Weight transpose + convert: W[K][N] fp32 -> Bt[N][K] bf16  (64x64 LDS tiles)
// ---------------------------------------------------------------------------
__global__ __launch_bounds__(256) void transpose_cvt(const float* __restrict__ W,
                                                     bf16* __restrict__ Bt,
                                                     int K, int N) {
    __shared__ float tile[64][65];
    const int t  = threadIdx.x;
    const int k0 = blockIdx.y * 64;
    const int n0 = blockIdx.x * 64;
#pragma unroll
    for (int i = 0; i < 4; ++i) {
        int q  = t + 256 * i;          // 0..1023 -> 64 rows x 16 float4 chunks
        int kk = q >> 4;
        int nc = (q & 15) * 4;
        float4 v = *reinterpret_cast<const float4*>(W + (size_t)(k0 + kk) * N + n0 + nc);
        tile[kk][nc + 0] = v.x; tile[kk][nc + 1] = v.y;
        tile[kk][nc + 2] = v.z; tile[kk][nc + 3] = v.w;
    }
    __syncthreads();
#pragma unroll
    for (int i = 0; i < 4; ++i) {
        int q  = t + 256 * i;
        int nn = q >> 4;
        int kc = (q & 15) * 4;
        v4bf o;
        o[0] = (bf16)tile[kc + 0][nn];
        o[1] = (bf16)tile[kc + 1][nn];
        o[2] = (bf16)tile[kc + 2][nn];
        o[3] = (bf16)tile[kc + 3][nn];
        *reinterpret_cast<v4bf*>(Bt + (size_t)(n0 + nn) * K + k0 + kc) = o;
    }
}

// ---------------------------------------------------------------------------
// Router: one wave per token. logits = x[t,:] @ rw[H,8]; softmax; top-4
// (lowest-index tie-break, matching jax.lax.top_k); renormalize; dense gates.
// ---------------------------------------------------------------------------
__global__ __launch_bounds__(256) void router_kernel(const float* __restrict__ x,
                                                     const float* __restrict__ rw,
                                                     float* __restrict__ gates) {
    const int wave = threadIdx.x >> 5;
    const int lane = threadIdx.x & 31;
    const int t = blockIdx.x * 8 + wave;
    if (t >= T_TOK) return;

    float p[NEXP];
#pragma unroll
    for (int e = 0; e < NEXP; ++e) p[e] = 0.f;

    const float* xr = x + (size_t)t * HDIM;
    for (int j = lane; j < HDIM; j += 32) {
        float xv = xr[j];
#pragma unroll
        for (int e = 0; e < NEXP; ++e) p[e] += xv * rw[j * NEXP + e];
    }
#pragma unroll
    for (int e = 0; e < NEXP; ++e)
#pragma unroll
        for (int off = 16; off > 0; off >>= 1) p[e] += __shfl_xor(p[e], off, 32);

    // all lanes hold the full logits; compute softmax + top-4 redundantly
    float mx = p[0];
#pragma unroll
    for (int e = 1; e < NEXP; ++e) mx = fmaxf(mx, p[e]);
    float sum = 0.f;
#pragma unroll
    for (int e = 0; e < NEXP; ++e) { p[e] = __expf(p[e] - mx); sum += p[e]; }
    float inv = 1.f / sum;
#pragma unroll
    for (int e = 0; e < NEXP; ++e) p[e] *= inv;

    float g[NEXP];
#pragma unroll
    for (int e = 0; e < NEXP; ++e) g[e] = 0.f;
    float ksum = 0.f;
#pragma unroll
    for (int k = 0; k < 4; ++k) {
        int bi = 0; float bv = -1.f;
#pragma unroll
        for (int e = 0; e < NEXP; ++e)
            if (p[e] > bv) { bv = p[e]; bi = e; }   // strict > keeps lowest index on ties
        g[bi] = bv; ksum += bv; p[bi] = -2.f;
    }
    float kinv = 1.f / ksum;
    if (lane == 0) {
#pragma unroll
        for (int e = 0; e < NEXP; ++e) gates[t * NEXP + e] = g[e] * kinv;
    }
}

// ---------------------------------------------------------------------------
// WMMA GEMM:  C = A[M,K](bf16) x Bt[N,K](bf16, pre-transposed)
//   MODE 0: C(bf16) = silu(A x B)
//   MODE 1: C(f32) += gates[m, expert] * (A x B)     (combine into d_out)
// Block tile 128x128, K-step 64 (two 16x16x32 sub-steps). 8 waves: 4(M) x 2(N);
// wave tile 32x64. Ping-pong LDS buffers, one barrier per K-step.
// ---------------------------------------------------------------------------
#define BM 128
#define BN 128
#define BK 64
#define LDK 72   // padded LDS stride: 144B rows, keeps b128 fragment loads aligned

template <int MODE>
__global__ __launch_bounds__(256) void moe_gemm(const bf16* __restrict__ A,
                                                const bf16* __restrict__ Bt,
                                                void* __restrict__ Cout,
                                                const float* __restrict__ gates,
                                                int N, int K, int expert) {
    __shared__ bf16 As[2][BM][LDK];
    __shared__ bf16 Bs[2][BN][LDK];

    const int tid   = threadIdx.x;
    const int lane  = tid & 31;
    const int wave  = tid >> 5;
    const int waveM = wave & 3;      // 4 waves along M
    const int waveN = wave >> 2;     // 2 waves along N
    const int bM = blockIdx.y * BM;
    const int bN = blockIdx.x * BN;

    // cooperative copy coords: each thread moves 64B of A and 64B of B per tile
    const int cRow = tid >> 1;           // 0..127
    const int cCol = (tid & 1) * 32;     // 0 or 32 (elements)

    const int laneM = lane & 15;
    const int aHalf = (lane >> 4) * 8;   // A frag: K {0-7,16-23} vs {8-15,24-31}
    const int bHalf = (lane >> 4) * 16;  // unused; B frag is read with aHalf (same K order as A)

    const bf16* gA = A  + (size_t)(bM + cRow) * K + cCol;
    const bf16* gB = Bt + (size_t)(bN + cRow) * K + cCol;

    v8f acc[2][4];
#pragma unroll
    for (int mi = 0; mi < 2; ++mi)
#pragma unroll
        for (int ni = 0; ni < 4; ++ni) acc[mi][ni] = (v8f){};

    // prologue: tile 0 -> buffer 0
    {
        bf16* la = &As[0][cRow][cCol];
        bf16* lb = &Bs[0][cRow][cCol];
#pragma unroll
        for (int c = 0; c < 4; ++c) {
            cp16(la + c * 8, gA + c * 8);
            cp16(lb + c * 8, gB + c * 8);
        }
    }
    cp_wait();
    __syncthreads();

    const int nk = K >> 6;           // K / BK
    for (int kt = 0; kt < nk; ++kt) {
        const int cur = kt & 1;
        const int nxt = cur ^ 1;

        // issue next tile's copy into the other buffer (no hazard: that buffer
        // was last read in the previous iteration, fenced by the barrier below)
        if (kt + 1 < nk) {
            const int k0 = (kt + 1) << 6;
            bf16* la = &As[nxt][cRow][cCol];
            bf16* lb = &Bs[nxt][cRow][cCol];
#pragma unroll
            for (int c = 0; c < 4; ++c) {
                cp16(la + c * 8, gA + k0 + c * 8);
                cp16(lb + c * 8, gB + k0 + c * 8);
            }
        }

        // compute from current buffer: 2 K-substeps of 32, 8 WMMAs each
#pragma unroll
        for (int ks = 0; ks < 2; ++ks) {
            const int kb = ks * 32;
            v16bf af[2], bfm[4];
#pragma unroll
            for (int mi = 0; mi < 2; ++mi) {
                int r = waveM * 32 + mi * 16 + laneM;
                v8bf lo = *reinterpret_cast<const v8bf*>(&As[cur][r][kb + aHalf]);
                v8bf hi = *reinterpret_cast<const v8bf*>(&As[cur][r][kb + 16 + aHalf]);
                af[mi] = join16(lo, hi);
            }
#pragma unroll
            for (int ni = 0; ni < 4; ++ni) {
                int c = waveN * 64 + ni * 16 + laneM;
                v8bf lo = *reinterpret_cast<const v8bf*>(&Bs[cur][c][kb + aHalf]);
                v8bf hi = *reinterpret_cast<const v8bf*>(&Bs[cur][c][kb + 16 + aHalf]);
                bfm[ni] = join16(lo, hi);
            }
#pragma unroll
            for (int mi = 0; mi < 2; ++mi)
#pragma unroll
                for (int ni = 0; ni < 4; ++ni)
                    acc[mi][ni] = __builtin_amdgcn_wmma_f32_16x16x32_bf16(
                        false, af[mi], false, bfm[ni], (short)0, acc[mi][ni], false, false);
        }

        cp_wait();         // next buffer's async writes have landed (this wave)
        __syncthreads();   // all waves: next buffer full, current buffer free
    }

    // Epilogue. C/D layout: lane%16 = N col; VGPR r -> M=r (lanes 0-15), r+8 (16-31)
    const int mHalf = (lane >> 4) * 8;
#pragma unroll
    for (int mi = 0; mi < 2; ++mi) {
#pragma unroll
        for (int r = 0; r < 8; ++r) {
            const int mRow = bM + waveM * 32 + mi * 16 + r + mHalf;
            float g = 0.f;
            if (MODE == 1) g = gates[mRow * NEXP + expert];
#pragma unroll
            for (int ni = 0; ni < 4; ++ni) {
                const int nCol = bN + waveN * 64 + ni * 16 + laneM;
                float v = acc[mi][ni][r];
                if (MODE == 0) {
                    float s = v / (1.f + __expf(-v));   // silu
                    reinterpret_cast<bf16*>(Cout)[(size_t)mRow * N + nCol] = (bf16)s;
                } else {
                    reinterpret_cast<float*>(Cout)[(size_t)mRow * N + nCol] += g * v;
                }
            }
        }
    }
}

// ---------------------------------------------------------------------------
// Host orchestration
// ---------------------------------------------------------------------------
static inline size_t alignUp(size_t x, size_t a) { return (x + a - 1) & ~(a - 1); }

extern "C" void kernel_launch(void* const* d_in, const int* in_sizes, int n_in,
                              void* d_out, int out_size, void* d_ws, size_t ws_size,
                              hipStream_t stream) {
    (void)in_sizes; (void)n_in; (void)out_size; (void)ws_size;
    const float* x  = (const float*)d_in[0];   // [T, H]
    const float* rw = (const float*)d_in[1];   // [H, 8]
    const float* w1 = (const float*)d_in[2];   // [E, H, I]
    const float* w2 = (const float*)d_in[3];   // [E, I, I]
    const float* w3 = (const float*)d_in[4];   // [E, I, H]
    float* out = (float*)d_out;                // [T, H]

    char* ws = (char*)d_ws;
    float* gates = (float*)ws; ws += alignUp((size_t)T_TOK * NEXP * sizeof(float), 256);
    bf16* xb = (bf16*)ws;      ws += alignUp((size_t)T_TOK * HDIM * sizeof(bf16), 256);
    bf16* h1 = (bf16*)ws;      ws += alignUp((size_t)T_TOK * IDIM * sizeof(bf16), 256);
    bf16* h2 = (bf16*)ws;      ws += alignUp((size_t)T_TOK * IDIM * sizeof(bf16), 256);
    bf16* wt = (bf16*)ws;      // reusable transposed-weight buffer, I*I bf16 (32 MB)

    const size_t n4 = (size_t)T_TOK * HDIM / 4;
    zero_f32<<<dim3(8192), 256, 0, stream>>>((float4*)out, n4);
    cvt_bf16<<<dim3(8192), 256, 0, stream>>>((const float4*)x, (v4bf*)xb, n4);
    router_kernel<<<dim3(T_TOK / 8), 256, 0, stream>>>(x, rw, gates);

    for (int e = 0; e < NEXP; ++e) {
        // layer 1: h1 = silu(x @ w1[e])          (K=H, N=I)
        transpose_cvt<<<dim3(IDIM / 64, HDIM / 64), 256, 0, stream>>>(
            w1 + (size_t)e * HDIM * IDIM, wt, HDIM, IDIM);
        moe_gemm<0><<<dim3(IDIM / BN, T_TOK / BM), 256, 0, stream>>>(
            xb, wt, h1, nullptr, IDIM, HDIM, e);
        // layer 2: h2 = silu(h1 @ w2[e])         (K=I, N=I)
        transpose_cvt<<<dim3(IDIM / 64, IDIM / 64), 256, 0, stream>>>(
            w2 + (size_t)e * IDIM * IDIM, wt, IDIM, IDIM);
        moe_gemm<0><<<dim3(IDIM / BN, T_TOK / BM), 256, 0, stream>>>(
            h1, wt, h2, nullptr, IDIM, IDIM, e);
        // layer 3: out += gate[:,e] * (h2 @ w3[e])  (K=I, N=H)
        transpose_cvt<<<dim3(HDIM / 64, IDIM / 64), 256, 0, stream>>>(
            w3 + (size_t)e * IDIM * HDIM, wt, IDIM, HDIM);
        moe_gemm<1><<<dim3(HDIM / BN, T_TOK / BM), 256, 0, stream>>>(
            h2, wt, out, gates, HDIM, IDIM, e);
    }
}
